// LTMHead_47931835023692
// MI455X (gfx1250) — matrix operand order repaired and checked by reading, not compiled, not run on hardware
//
#include <hip/hip_runtime.h>

typedef __bf16 bf16;
typedef __attribute__((ext_vector_type(16))) __bf16 bf16x16;
typedef __attribute__((ext_vector_type(8)))  float  f32x8;

constexpr int B_ = 8, T_ = 1024, M_ = 8192, E_ = 1024, H_ = 128;
constexpr int S_ = M_ + T_;          // 9216
constexpr int PTS_ = 32;
constexpr int NS_ = S_ / 16;         // 576 s-tiles
constexpr int NT_ = T_ / 16;         // 64  t-tiles
constexpr int NH_ = H_ / 16;         // 8   h-tiles
constexpr int SGRP_ = 4;             // s-tiles per wave in scores kernel
constexpr int NSG_ = NS_ / SGRP_;    // 144
constexpr int HG_ = 4;               // h-tiles per wave in out kernel (2 groups)
constexpr float DECAY_ = 0.01f, LQ_ADD_ = 1e-4f;

// d_out layout (float elements): out | qt_loss | new_memory | new_dist | new_rank
constexpr size_t OUT_OFF = 0;
constexpr size_t QT_OFF  = (size_t)B_ * T_ * H_;
constexpr size_t NM_OFF  = QT_OFF + 1;
constexpr size_t ND_OFF  = NM_OFF + (size_t)B_ * M_ * E_;
constexpr size_t NR_OFF  = ND_OFF + (size_t)B_ * M_;

// ---------------- fragment helpers (CDNA5 WMMA wave32 layouts) ----------------

__device__ __forceinline__ bf16 f2bf(float f) {
    unsigned int u = __float_as_uint(f);
    unsigned int r = u + 0x7FFFu + ((u >> 16) & 1u);   // round-nearest-even
    unsigned short h = (unsigned short)(r >> 16);
    bf16 o;
    __builtin_memcpy(&o, &h, 2);
    return o;
}

// A: 16x32 (MxK) bf16, row-major source, lda elements.
// lane L<16: row=L, K=0..15; lane L>=16: row=L-16, K=16..31
__device__ __forceinline__ bf16x16 load_a(const bf16* src, int lda) {
    int lane = threadIdx.x & 31;
    int m = lane & 15;
    int kb = (lane < 16) ? 0 : 16;
    const bf16* p = src + (size_t)m * lda + kb;
    bf16x16 a;
    __builtin_memcpy(&a, p, 32);
    return a;
}

// B: 32x16 (KxN) bf16 supplied "N-major": row n holds K contiguous (ldb elements).
// lane L<16: col=L, K=0..15; lane L>=16: col=L-16, K=16..31  -> one 32B run per lane
__device__ __forceinline__ bf16x16 load_b(const bf16* srcN, int ldb) {
    int lane = threadIdx.x & 31;
    int n = lane & 15;
    int kb = (lane < 16) ? 0 : 16;
    const bf16* p = srcN + (size_t)n * ldb + kb;
    bf16x16 b;
    __builtin_memcpy(&b, p, 32);
    return b;
}

__device__ __forceinline__ f32x8 wmma_bf16(bf16x16 a, bf16x16 b, f32x8 c) {
    return __builtin_amdgcn_wmma_f32_16x16x32_bf16(false, a, false, b, (short)0, c,
                                                   false, false);
}

// C/D 16x16 f32: VGPR r, lanes 0-15 -> (M=r, N=lane); lanes 16-31 -> (M=8+r, N=lane-16)
__device__ __forceinline__ void store_c_f32(float* dst, int ldd, f32x8 c) {
    int lane = threadIdx.x & 31;
    int n = lane & 15, mo = (lane >> 4) * 8;
#pragma unroll
    for (int r = 0; r < 8; r++) dst[(size_t)(mo + r) * ldd + n] = c[r];
}
__device__ __forceinline__ void store_c_bf(bf16* dst, int ldd, f32x8 c) {
    int lane = threadIdx.x & 31;
    int n = lane & 15, mo = (lane >> 4) * 8;
#pragma unroll
    for (int r = 0; r < 8; r++) dst[(size_t)(mo + r) * ldd + n] = f2bf(c[r]);
}
__device__ __forceinline__ void store_c_bf_T(bf16* dst, int ldd, f32x8 c) {  // dst[n][m]
    int lane = threadIdx.x & 31;
    int n = lane & 15, mo = (lane >> 4) * 8;
#pragma unroll
    for (int r = 0; r < 8; r++) dst[(size_t)n * ldd + mo + r] = f2bf(c[r]);
}

// ---------------- kernels ----------------

__global__ __launch_bounds__(256) void convw_kernel(const float* Wk, const float* Wq,
                                                    const float* Wv, bf16* wkb,
                                                    bf16* wqb, bf16* wvb) {
    int i = blockIdx.x * 256 + threadIdx.x;
    if (i < H_ * E_) {
        wkb[i] = f2bf(Wk[i]);
        wqb[i] = f2bf(Wq[i]);
        wvb[i] = f2bf(Wv[i]);
    }
}

// build mk = [memory(reset)+pos_emb ; inp] in bf16; record dist(int)
__global__ __launch_bounds__(256) void prep_kernel(const int* bpl, const float* inp,
                                                   const float* memory, const int* mbd,
                                                   const float* pos_table, bf16* mk,
                                                   int* distArr) {
    int row = blockIdx.x;                 // b*S + s
    int b = row / S_, s = row - b * S_;
    int tid = threadIdx.x;
    bool reset = (bpl[b] == 0);
    bf16* orow = mk + ((size_t)b * S_ + s) * E_;
    if (s < M_) {
        int d = (reset ? 0 : mbd[b * M_ + s]) + 1;
        int ix = 31 - __clz(d);
        if (ix > PTS_ - 1) ix = PTS_ - 1;
        if (tid == 0) distArr[b * M_ + s] = d;
        const float* mrow = memory + ((size_t)b * M_ + s) * E_;
        const float* prow = pos_table + (size_t)ix * E_;
        for (int e = tid; e < E_; e += 256)
            orow[e] = f2bf((reset ? 0.f : mrow[e]) + prow[e]);
    } else {
        const float* irow = inp + ((size_t)b * T_ + (s - M_)) * E_;
        for (int e = tid; e < E_; e += 256) orow[e] = f2bf(irow[e]);
    }
}

// k = mk Wk^T, v = mk Wv^T (v stored transposed). One wave per (b, s-tile) computes
// the FULL H strip (8 h-tiles) for both k and v, so the big streamed operand (mk)
// is loaded exactly once and one A fragment feeds 16 WMMAs per K-step.
__global__ __launch_bounds__(256) void gemm_kv_kernel(const bf16* mk, const bf16* wkb,
                                                      const bf16* wvb, bf16* kb,
                                                      bf16* vt) {
    int wave = (blockIdx.x * 256 + threadIdx.x) >> 5;
    int st = wave % NS_, b = wave / NS_;
    if (b >= B_) return;
    int s0 = st * 16;
    const bf16* Abase = mk + ((size_t)b * S_ + s0) * E_;
    f32x8 ck[NH_] = {};
    f32x8 cv[NH_] = {};
    for (int e0 = 0; e0 < E_; e0 += 32) {
        bf16x16 a = load_a(Abase + e0, E_);
#pragma unroll
        for (int h = 0; h < NH_; h++) {
            ck[h] = wmma_bf16(a, load_b(wkb + (size_t)(h * 16) * E_ + e0, E_), ck[h]);
            cv[h] = wmma_bf16(a, load_b(wvb + (size_t)(h * 16) * E_ + e0, E_), cv[h]);
        }
    }
#pragma unroll
    for (int h = 0; h < NH_; h++) {
        store_c_bf(kb + ((size_t)b * S_ + s0) * H_ + h * 16, H_, ck[h]);
        store_c_bf_T(vt + ((size_t)b * H_ + h * 16) * S_ + s0, S_, cv[h]);
    }
}

// q = inp Wq^T over the inp rows of mk; one wave per (b, t-tile), full H strip.
__global__ __launch_bounds__(256) void gemm_q_kernel(const bf16* mk, const bf16* wqb,
                                                     bf16* qb) {
    int wave = (blockIdx.x * 256 + threadIdx.x) >> 5;
    int tt = wave % NT_, b = wave / NT_;
    if (b >= B_) return;
    int t0 = tt * 16;
    const bf16* Abase = mk + ((size_t)b * S_ + M_ + t0) * E_;
    f32x8 cq[NH_] = {};
    for (int e0 = 0; e0 < E_; e0 += 32) {
        bf16x16 a = load_a(Abase + e0, E_);
#pragma unroll
        for (int h = 0; h < NH_; h++)
            cq[h] = wmma_bf16(a, load_b(wqb + (size_t)(h * 16) * E_ + e0, E_), cq[h]);
    }
#pragma unroll
    for (int h = 0; h < NH_; h++)
        store_c_bf(qb + ((size_t)b * T_ + t0) * H_ + h * 16, H_, cq[h]);
}

// raw = (q k^T)*scale (f32); rowmax = max_s raw^2 via uint-bits atomicMax (vals >= 0).
// One wave per (b, t-tile, group of 4 s-tiles): A (q) fragment feeds 4 WMMAs per step.
__global__ __launch_bounds__(256) void scores_kernel(const bf16* qb, const bf16* kb,
                                                     float* raw, unsigned int* rowmax) {
    int wave = (blockIdx.x * 256 + threadIdx.x) >> 5;
    int sg = wave % NSG_;
    int tmp = wave / NSG_;
    int tt = tmp % NT_, b = tmp / NT_;
    if (b >= B_) return;
    int t0 = tt * 16, s0 = sg * SGRP_ * 16;
    const bf16* A = qb + ((size_t)b * T_ + t0) * H_;
    f32x8 c[SGRP_] = {};
#pragma unroll
    for (int h0 = 0; h0 < H_; h0 += 32) {
        bf16x16 a = load_a(A + h0, H_);
#pragma unroll
        for (int i = 0; i < SGRP_; i++)
            c[i] = wmma_bf16(a, load_b(kb + ((size_t)b * S_ + s0 + i * 16) * H_ + h0, H_),
                             c[i]);
    }
    const float scale = rsqrtf((float)S_);
    int lane = threadIdx.x & 31;
    int n = lane & 15, mo = (lane >> 4) * 8;
    unsigned int* rm = rowmax + b * T_ + t0;
#pragma unroll
    for (int i = 0; i < SGRP_; i++) {
        float* dst = raw + ((size_t)b * T_ + t0) * S_ + s0 + i * 16;
#pragma unroll
        for (int r = 0; r < 8; r++) {
            float v = c[i][r] * scale;
            dst[(size_t)(mo + r) * S_ + n] = v;
            atomicMax(&rm[mo + r], __float_as_uint(v * v));
        }
    }
}

// wei = raw^2 / rowmax -> bf16; colsum += wei over t; qt_loss partial
__global__ __launch_bounds__(256) void norm_kernel(const float* raw,
                                                   const unsigned int* rowmax,
                                                   bf16* weib, float* colsum,
                                                   float* qtl) {
    constexpr int SC = S_ / 256;  // 36
    int sc = blockIdx.x % SC;
    int tmp = blockIdx.x / SC;
    int tt = tmp % NT_, b = tmp / NT_;
    int s = sc * 256 + threadIdx.x;
    int t0 = tt * 16;
    const float* rbase = raw + ((size_t)b * T_ + t0) * S_ + s;
    bf16* wbase = weib + ((size_t)b * T_ + t0) * S_ + s;
    float cacc = 0.f, lacc = 0.f;
#pragma unroll
    for (int r = 0; r < 16; r++) {
        float rm = fmaxf(__uint_as_float(rowmax[b * T_ + t0 + r]), 1e-30f);
        float v = rbase[(size_t)r * S_];
        float w = v * v / rm;
        wbase[(size_t)r * S_] = f2bf(w);
        cacc += w;
        lacc += __logf(w + LQ_ADD_);
    }
    atomicAdd(&colsum[b * S_ + s], cacc);
    __shared__ float red[256];
    red[threadIdx.x] = lacc;
    __syncthreads();
    for (int o = 128; o > 0; o >>= 1) {
        if (threadIdx.x < o) red[threadIdx.x] += red[threadIdx.x + o];
        __syncthreads();
    }
    if (threadIdx.x == 0) atomicAdd(qtl, red[0]);
}

// out = wei v. One wave per (b, t-tile, half-H): A (wei) fragment feeds 4 WMMAs per
// step, so the 151MB wei buffer is streamed 2x instead of 8x.
__global__ __launch_bounds__(256) void outgemm_kernel(const bf16* weib, const bf16* vt,
                                                      float* out) {
    int wave = (blockIdx.x * 256 + threadIdx.x) >> 5;
    int hg = wave & 1;
    int tmp = wave >> 1;
    int tt = tmp % NT_, b = tmp / NT_;
    if (b >= B_) return;
    int t0 = tt * 16, h0 = hg * HG_ * 16;
    const bf16* A = weib + ((size_t)b * T_ + t0) * S_;
    f32x8 c[HG_] = {};
    for (int s0 = 0; s0 < S_; s0 += 32) {
        bf16x16 a = load_a(A + s0, S_);
#pragma unroll
        for (int i = 0; i < HG_; i++)
            c[i] = wmma_bf16(
                a, load_b(vt + ((size_t)b * H_ + h0 + i * 16) * S_ + s0, S_), c[i]);
    }
#pragma unroll
    for (int i = 0; i < HG_; i++)
        store_c_f32(out + OUT_OFF + ((size_t)b * T_ + t0) * H_ + h0 + i * 16, H_, c[i]);
}

// mean_wei + new_rank
__global__ __launch_bounds__(256) void rank_kernel(const float* colsum, const float* mrank,
                                                   float* rank) {
    int b = blockIdx.x;
    __shared__ float red[256];
    float acc = 0.f;
    for (int s = threadIdx.x; s < S_; s += 256) acc += colsum[b * S_ + s];
    red[threadIdx.x] = acc;
    __syncthreads();
    for (int o = 128; o > 0; o >>= 1) {
        if (threadIdx.x < o) red[threadIdx.x] += red[threadIdx.x + o];
        __syncthreads();
    }
    float mw = red[0] / (float)S_;
    for (int m = threadIdx.x; m < M_; m += 256)
        rank[b * M_ + m] = mrank[b * M_ + m] + colsum[b * S_ + m] / mw - 1.0f - DECAY_;
}

// per-batch bitonic sort of (rank, idx) descending (sort ascending on -rank) in LDS
__global__ __launch_bounds__(1024) void sort_kernel(const float* rank, float* rankS,
                                                    int* ordS) {
    __shared__ float sk[M_];
    __shared__ int si[M_];
    int b = blockIdx.x;
    for (int i = threadIdx.x; i < M_; i += 1024) {
        sk[i] = -rank[b * M_ + i];
        si[i] = i;
    }
    __syncthreads();
    for (int k = 2; k <= M_; k <<= 1) {
        for (int j = k >> 1; j > 0; j >>= 1) {
            for (int i = threadIdx.x; i < M_; i += 1024) {
                int ixj = i ^ j;
                if (ixj > i) {
                    bool up = ((i & k) == 0);
                    float a = sk[i], c = sk[ixj];
                    if ((up && a > c) || (!up && a < c)) {
                        sk[i] = c;
                        sk[ixj] = a;
                        int t = si[i];
                        si[i] = si[ixj];
                        si[ixj] = t;
                    }
                }
            }
            __syncthreads();
        }
    }
    for (int i = threadIdx.x; i < M_; i += 1024) {
        rankS[b * M_ + i] = -sk[i];
        ordS[b * M_ + i] = si[i];
    }
}

// gather sorted memory, evict bottom r -> inp tail; write dist/rank/qt_loss
__global__ __launch_bounds__(256) void gather_kernel(const int* bpl, const float* memory,
                                                     const float* inp, const int* distArr,
                                                     const float* rankS, const int* ordS,
                                                     const float* qtl, float* out) {
    constexpr int r = (T_ < M_) ? T_ : M_;
    int row = blockIdx.x;  // b*M + i
    int b = row / M_, i = row - b * M_;
    bool reset = (bpl[b] == 0);
    float* nm = out + NM_OFF + ((size_t)b * M_ + i) * E_;
    if (i < M_ - r) {
        int src = ordS[b * M_ + i];
        const float* srow = memory + ((size_t)b * M_ + src) * E_;
        for (int e = threadIdx.x; e < E_; e += 256) nm[e] = reset ? 0.f : srow[e];
        if (threadIdx.x == 0) {
            ((int*)out)[ND_OFF + b * M_ + i] = distArr[b * M_ + src];
            out[NR_OFF + b * M_ + i] = rankS[b * M_ + i];
        }
    } else {
        int t = (T_ - r) + (i - (M_ - r));
        const float* srow = inp + ((size_t)b * T_ + t) * E_;
        for (int e = threadIdx.x; e < E_; e += 256) nm[e] = srow[e];
        if (threadIdx.x == 0) {
            ((int*)out)[ND_OFF + b * M_ + i] = 0;
            out[NR_OFF + b * M_ + i] = 1.0f;
        }
    }
    if (row == 0 && threadIdx.x == 1) out[QT_OFF] = qtl[0];
}

// ---------------- host ----------------

extern "C" void kernel_launch(void* const* d_in, const int* in_sizes, int n_in,
                              void* d_out, int out_size, void* d_ws, size_t ws_size,
                              hipStream_t stream) {
    (void)in_sizes; (void)n_in; (void)out_size; (void)ws_size;
    const int*   bpl = (const int*)d_in[0];
    const float* inp = (const float*)d_in[1];
    const float* mem = (const float*)d_in[2];
    const int*   mbd = (const int*)d_in[3];
    const float* mrk = (const float*)d_in[4];
    const float* pos = (const float*)d_in[5];
    const float* Wk  = (const float*)d_in[6];
    const float* Wq  = (const float*)d_in[7];
    const float* Wv  = (const float*)d_in[8];
    float* out = (float*)d_out;
    char* ws = (char*)d_ws;

    size_t o = 0;
    auto alloc = [&](size_t n) { size_t r = o; o += (n + 255) & ~(size_t)255; return r; };
    size_t MKo  = alloc((size_t)B_ * S_ * E_ * 2);
    size_t WKo  = alloc((size_t)H_ * E_ * 2);
    size_t WQo  = alloc((size_t)H_ * E_ * 2);
    size_t WVo  = alloc((size_t)H_ * E_ * 2);
    size_t KBo  = alloc((size_t)B_ * S_ * H_ * 2);
    size_t VTo  = alloc((size_t)B_ * H_ * S_ * 2);
    size_t QBo  = alloc((size_t)B_ * T_ * H_ * 2);
    size_t RAWo = alloc((size_t)B_ * T_ * S_ * 4);
    size_t WEIo = alloc((size_t)B_ * T_ * S_ * 2);
    size_t RMo  = alloc((size_t)B_ * T_ * 4);
    size_t CSo  = alloc((size_t)B_ * S_ * 4);
    size_t QTo  = alloc(256);
    size_t RKo  = alloc((size_t)B_ * M_ * 4);
    size_t RSo  = alloc((size_t)B_ * M_ * 4);
    size_t ORo  = alloc((size_t)B_ * M_ * 4);
    size_t DIo  = alloc((size_t)B_ * M_ * 4);

    hipMemsetAsync(ws + RMo, 0, (size_t)B_ * T_ * 4, stream);
    hipMemsetAsync(ws + CSo, 0, (size_t)B_ * S_ * 4, stream);
    hipMemsetAsync(ws + QTo, 0, 4, stream);

    convw_kernel<<<(H_ * E_ + 255) / 256, 256, 0, stream>>>(
        Wk, Wq, Wv, (bf16*)(ws + WKo), (bf16*)(ws + WQo), (bf16*)(ws + WVo));
    prep_kernel<<<B_ * S_, 256, 0, stream>>>(bpl, inp, mem, mbd, pos,
                                             (bf16*)(ws + MKo), (int*)(ws + DIo));
    gemm_kv_kernel<<<B_ * NS_ / 8, 256, 0, stream>>>(
        (const bf16*)(ws + MKo), (const bf16*)(ws + WKo), (const bf16*)(ws + WVo),
        (bf16*)(ws + KBo), (bf16*)(ws + VTo));
    gemm_q_kernel<<<B_ * NT_ / 8, 256, 0, stream>>>(
        (const bf16*)(ws + MKo), (const bf16*)(ws + WQo), (bf16*)(ws + QBo));
    scores_kernel<<<B_ * NT_ * NSG_ / 8, 256, 0, stream>>>(
        (const bf16*)(ws + QBo), (const bf16*)(ws + KBo), (float*)(ws + RAWo),
        (unsigned int*)(ws + RMo));
    norm_kernel<<<B_ * NT_ * (S_ / 256), 256, 0, stream>>>(
        (const float*)(ws + RAWo), (const unsigned int*)(ws + RMo),
        (bf16*)(ws + WEIo), (float*)(ws + CSo), (float*)(ws + QTo));
    outgemm_kernel<<<B_ * NT_ * 2 / 8, 256, 0, stream>>>(
        (const bf16*)(ws + WEIo), (const bf16*)(ws + VTo), out);
    rank_kernel<<<B_, 256, 0, stream>>>((const float*)(ws + CSo), mrk,
                                        (float*)(ws + RKo));
    sort_kernel<<<B_, 1024, 0, stream>>>((const float*)(ws + RKo),
                                         (float*)(ws + RSo), (int*)(ws + ORo));
    gather_kernel<<<B_ * M_, 256, 0, stream>>>(bpl, mem, inp, (const int*)(ws + DIo),
                                               (const float*)(ws + RSo),
                                               (const int*)(ws + ORo),
                                               (const float*)(ws + QTo), out);
}
